// CoOccurrenceMatrixFast_1297080123419
// MI455X (gfx1250) — matrix operand order repaired and checked by reading, not compiled, not run on hardware
//
#include <hip/hip_runtime.h>
#include <math.h>

typedef __attribute__((ext_vector_type(16))) _Float16 v16h;
typedef __attribute__((ext_vector_type(8)))  _Float16 v8h;
typedef __attribute__((ext_vector_type(8)))  float    v8f;
typedef __attribute__((ext_vector_type(16))) float    v16f;

#define KSPLIT 128   // FC1 split-K groups (4096 k-chunks / 128 = 32 chunks each)

// ------------------------------------------------------------------
// Co-occurrence: one workgroup per (b,c) plane. 256KB LDS histogram
// (CDNA5: up to 320KB LDS per workgroup). ds_add_u32 atomics, then
// one f16 store of the 256x256 count matrix (planar layout).
// ------------------------------------------------------------------
__global__ __launch_bounds__(1024)
void cooc_kernel(const float* __restrict__ x, _Float16* __restrict__ out) {
    extern __shared__ unsigned int hist[];              // 65536 u32 = 256KB
    const int plane = blockIdx.x;                       // b*3 + c
    const int c = plane % 3;
    const float mean_c = (c == 0) ? 0.485f : (c == 1) ? 0.456f : 0.406f;
    const float std_c  = (c == 0) ? 0.229f : (c == 1) ? 0.224f : 0.225f;
    const float* xp = x + (long long)plane * 65536;
    const int tid = threadIdx.x;

    for (int i = tid; i < 65536; i += 1024) hist[i] = 0u;
    __syncthreads();

    for (int i = tid; i < 65536; i += 1024) {
        const int yk = i >> 8, xk = i & 255;
        float v00 = fminf(fmaxf(xp[i] * std_c + mean_c, 0.f), 1.f);
        int q00 = (int)(v00 * 255.0f);
        if (xk < 255) {
            float v01 = fminf(fmaxf(xp[i + 1] * std_c + mean_c, 0.f), 1.f);
            int q01 = (int)(v01 * 255.0f);
            atomicAdd(&hist[q00 * 256 + q01], 1u);                 // horizontal
            if (yk < 255) {
                float v10 = fminf(fmaxf(xp[i + 256] * std_c + mean_c, 0.f), 1.f);
                float v11 = fminf(fmaxf(xp[i + 257] * std_c + mean_c, 0.f), 1.f);
                int q10 = (int)(v10 * 255.0f);
                int q11 = (int)(v11 * 255.0f);
                atomicAdd(&hist[q00 * 256 + q10], 1u);             // vertical
                atomicAdd(&hist[q00 * 256 + q11], 1u);             // diag 45
                atomicAdd(&hist[q01 * 256 + q10], 1u);             // diag 135
            }
        } else if (yk < 255) {
            float v10 = fminf(fmaxf(xp[i + 256] * std_c + mean_c, 0.f), 1.f);
            int q10 = (int)(v10 * 255.0f);
            atomicAdd(&hist[q00 * 256 + q10], 1u);                 // vertical
        }
    }
    __syncthreads();
    for (int i = tid; i < 65536; i += 1024)
        out[(long long)plane * 65536 + i] = (_Float16)(float)hist[i];
}

// ------------------------------------------------------------------
// Conv1: 3->32, 3x3, pad1, relu. Cin=3 => 27 MACs/output; VALU direct.
// Input planar (co-occurrence), output 32-channel-blocked.
// ------------------------------------------------------------------
__global__ __launch_bounds__(256)
void conv1_kernel(const _Float16* __restrict__ in, const float* __restrict__ w,
                  const float* __restrict__ bias, _Float16* __restrict__ out) {
    const int co = threadIdx.x & 31;
    const int xi = threadIdx.x >> 5;                    // 0..7
    const int x  = blockIdx.x * 8 + xi;
    const int y  = blockIdx.y;
    const int b  = blockIdx.z;
    float acc = bias[co];
    for (int c = 0; c < 3; ++c)
        for (int ky = 0; ky < 3; ++ky) {
            const int yy = y + ky - 1;
            if (yy < 0 || yy > 255) continue;
            const _Float16* row = in + ((long long)(b * 3 + c) * 256 + yy) * 256;
            for (int kx = 0; kx < 3; ++kx) {
                const int xx = x + kx - 1;
                if (xx < 0 || xx > 255) continue;
                acc += (float)row[xx] * w[((co * 3 + c) * 3 + ky) * 3 + kx];
            }
        }
    acc = fmaxf(acc, 0.f);
    out[(((long long)b * 256 + y) * 256 + x) * 32 + co] = (_Float16)acc;
}

// ------------------------------------------------------------------
// Pack f32 OIHW weights into WMMA A-fragment order (f16):
// dst layout [cot][ky][cb][kx][lane(32)][16 halfs in per-lane k order].
// ------------------------------------------------------------------
__global__ __launch_bounds__(256)
void pack_weights(const float* __restrict__ w, _Float16* __restrict__ dst,
                  int CO, int CI, int KS, int total) {
    int idx = blockIdx.x * 256 + threadIdx.x;
    if (idx >= total) return;
    const int h    = idx & 15;
    const int lane = (idx >> 4) & 31;
    int t = idx >> 9;
    const int kx = t % KS;           t /= KS;
    const int cb = t % (CI >> 5);    t /= (CI >> 5);
    const int ky = t % KS;           t /= KS;
    const int cot = t;
    const int g = lane >> 4, m = lane & 15;
    const int j = h >> 1, p = h & 1;
    const int k  = 16 * g + h;
    const int ci = cb * 32 + k;
    const int co = cot * 16 + m;
    dst[idx] = (_Float16)w[((co * CI + ci) * KS + ky) * KS + kx];
}

// ------------------------------------------------------------------
// WMMA implicit-GEMM conv, 32-channel-blocked activations, with
// DOUBLE-BUFFERED async global->LDS staging: the async batch for tile
// t+1 (3 x global_load_async_to_lds_b128) is issued before the wave
// waits on tile t (s_wait_asynccnt 0x3 -- async loads retire in order,
// so <=3 outstanding proves the older batch of 3 has landed). The copy
// latency for the next tile hides behind the current tile's WMMAs.
// s_wait_dscnt 0 closes the LDS WAR window (fragment ds reads vs the
// async engine's writes, which use a different counter).
// ------------------------------------------------------------------
template<int CI, int CO, int KS, int PAD, bool RELU>
__global__ __launch_bounds__(32)
void conv_wmma(const _Float16* __restrict__ in, const _Float16* __restrict__ wpack,
               const float* __restrict__ bias, _Float16* __restrict__ out,
               int H, int W) {
    constexpr int WT = 16 + 2 * PAD;                    // staged x-width
    constexpr int CHUNKS = WT * 4;                      // 16B chunks in tile
    constexpr int CB = CI / 32;
    constexpr int NT = KS * CB;                         // tiles (ky,cb)
    __shared__ __attribute__((aligned(16))) _Float16 sh[2][WT * 32];

    const int lane   = threadIdx.x;
    const int xtiles = W >> 4;
    const int xt  = blockIdx.x % xtiles;
    const int cot = blockIdx.x / xtiles;
    const int y   = blockIdx.y;
    const int b   = blockIdx.z;
    const int x0  = xt << 4;
    const int g   = lane >> 4;
    const int lm  = lane & 15;
    const unsigned lds0 = (unsigned)(size_t)(&sh[0][0]);  // low 32 bits = LDS offset

    auto issue_tile = [&](int t, int bufIdx) {
        const int ky = t / CB, cb = t % CB;
        const int yy = y + ky - PAD;
        const bool yin = (yy >= 0) && (yy < H);
        const long long plane = (long long)(b * CB + cb) * H * W * 32;
        const long long rowb  = plane + ((long long)yy * W + (x0 - PAD)) * 32;
        const unsigned lbase = lds0 + (unsigned)bufIdx * (unsigned)(WT * 64);
#pragma unroll
        for (int it = 0; it < 3; ++it) {                // exactly 3 async instrs
            const int q = lane + it * 32;
            if (q < CHUNKS) {
                const int xg = x0 - PAD + (q >> 2);
                const bool ok = yin && (xg >= 0) && (xg < W);
                const _Float16* gp = ok ? (in + rowb + (long long)q * 8)
                                        : (in + plane);        // safe dummy
                asm volatile("global_load_async_to_lds_b128 %0, %1, off"
                             :: "v"(lbase + (unsigned)q * 16u),
                                "v"((unsigned long long)(size_t)gp)
                             : "memory");
            }
        }
    };
    auto fix_tile = [&](int t, int bufIdx) {            // zero OOB 16B chunks
        const int ky = t / CB;
        const int yy = y + ky - PAD;
        const bool yin = (yy >= 0) && (yy < H);
#pragma unroll
        for (int it = 0; it < 3; ++it) {
            const int q = lane + it * 32;
            if (q < CHUNKS) {
                const int xg = x0 - PAD + (q >> 2);
                if (!(yin && (xg >= 0) && (xg < W))) {
                    v8h z = {};
                    *reinterpret_cast<v8h*>(&sh[bufIdx][q * 8]) = z;
                }
            }
        }
    };

    v8f acc = {};
    issue_tile(0, 0);
    for (int t = 0; t < NT; ++t) {
        const int cur = t & 1;
        if (t + 1 < NT) {
            // WAR: fragment ds reads that last touched buf cur^1 must retire
            asm volatile("s_wait_dscnt 0x0" ::: "memory");
            issue_tile(t + 1, cur ^ 1);
            asm volatile("s_wait_asynccnt 0x3" ::: "memory");   // tile t landed
        } else {
            asm volatile("s_wait_asynccnt 0x0" ::: "memory");
        }
        fix_tile(t, cur);
        const int ky = t / CB, cb = t % CB;
        const long long abase =
            ((((long long)cot * KS + ky) * CB + cb) * KS) * 512;
        for (int kx = 0; kx < KS; ++kx) {
            v16h a = *reinterpret_cast<const v16h*>(
                wpack + abase + (long long)(kx * 32 + lane) * 16);
            v16h bb = *reinterpret_cast<const v16h*>(&sh[cur][(lm + kx) * 32 + g * 16]);
            acc = __builtin_amdgcn_wmma_f32_16x16x32_f16(
                false, a, false, bb, (short)0, acc, false, false);
        }
    }
    // Epilogue: C layout M = 8g + r, N = lm. 8 consecutive channels in the
    // blocked layout -> one 16B store per lane.
    const int xo  = x0 + lm;
    const int cob = cot >> 1;
    const int ch0 = (cot & 1) * 16 + g * 8;
    v8h ov;
    for (int r = 0; r < 8; ++r) {
        float v = acc[r] + bias[cot * 16 + g * 8 + r];
        if (RELU) v = fmaxf(v, 0.f);
        ov[r] = (_Float16)v;
    }
    *reinterpret_cast<v8h*>(
        out + ((((long long)b * (CO / 32) + cob) * H + y) * W + xo) * 32 + ch0) = ov;
}

// ------------------------------------------------------------------
// 2x2 stride-2 maxpool on blocked layout. PLANAR_OUT de-blocks to
// planar NCHW (used before the FC flatten).
// ------------------------------------------------------------------
template<bool PLANAR_OUT>
__global__ __launch_bounds__(256)
void maxpool2_blk(const _Float16* __restrict__ in, _Float16* __restrict__ out,
                  int CB, int H, int W, int total) {
    int i = blockIdx.x * 256 + threadIdx.x;
    if (i >= total) return;
    const int c = i & 31;
    int t = i >> 5;
    const int Wo = W >> 1, Ho = H >> 1;
    const int xo = t % Wo; t /= Wo;
    const int yo = t % Ho; t /= Ho;
    const int p  = t;                                   // b*CB + cb
    const long long rb = (((long long)p * H + yo * 2) * W + xo * 2) * 32 + c;
    const long long rs = (long long)W * 32;
    float m0 = fmaxf((float)in[rb], (float)in[rb + 32]);
    float m1 = fmaxf((float)in[rb + rs], (float)in[rb + rs + 32]);
    const float m = fmaxf(m0, m1);
    if (PLANAR_OUT) {
        const int b = p / CB, cb = p % CB;
        const int cg = cb * 32 + c;
        out[(((long long)b * (CB * 32) + cg) * Ho + yo) * Wo + xo] = (_Float16)m;
    } else {
        out[(((long long)p * Ho + yo) * Wo + xo) * 32 + c] = (_Float16)m;
    }
}

__global__ __launch_bounds__(256)
void zero_f32(float* p, int n) {
    int i = blockIdx.x * 256 + threadIdx.x;
    if (i < n) p[i] = 0.f;
}

// ------------------------------------------------------------------
// FC1: [16 x 131072] @ fw1^T -> [16 x 256]. Split-K WMMA, f32 atomic
// reduction. Reads the 134MB f32 weight exactly once, cvt to f16 inline.
// ------------------------------------------------------------------
__global__ __launch_bounds__(32)
void fc1_wmma(const _Float16* __restrict__ h, const float* __restrict__ fw,
              float* __restrict__ accum) {
    const int lane = threadIdx.x;
    const int nt = blockIdx.x;                          // 0..15 (cout tile)
    const int ks = blockIdx.y;                          // 0..KSPLIT-1
    const int g = lane >> 4, lm = lane & 15;
    const long long K = 131072;
    const _Float16* ap = h + (long long)lm * K;         // A row m = lm
    const float* bp = fw + (long long)(nt * 16 + lm) * K; // B col n = lm

    v8f acc = {};
    for (int c = 0; c < 4096 / KSPLIT; ++c) {
        const int kb = (ks * (4096 / KSPLIT) + c) * 32;
        v8h alo = *reinterpret_cast<const v8h*>(ap + kb + 16 * g);
        v8h ahi = *reinterpret_cast<const v8h*>(ap + kb + 16 * g + 8);
        v16h a;
        for (int i = 0; i < 8; ++i) { a[i] = alo[i]; a[8 + i] = ahi[i]; }
        v16f bf = *reinterpret_cast<const v16f*>(bp + kb + 16 * g);
        v16h bb;
        for (int i = 0; i < 16; ++i) bb[i] = (_Float16)bf[i];
        acc = __builtin_amdgcn_wmma_f32_16x16x32_f16(
            false, a, false, bb, (short)0, acc, false, false);
    }
    for (int r = 0; r < 8; ++r) {
        const int m = g * 8 + r;                        // batch row
        atomicAdd(&accum[m * 256 + nt * 16 + lm], acc[r]);
    }
}

__global__ __launch_bounds__(256)
void fc1_post(const float* __restrict__ accum, const float* __restrict__ bias,
              _Float16* __restrict__ h1) {
    int i = blockIdx.x * 256 + threadIdx.x;
    if (i >= 16 * 256) return;
    h1[i] = (_Float16)fmaxf(accum[i] + bias[i & 255], 0.f);
}

// ------------------------------------------------------------------
// FC2: [16 x 256] @ fw2^T + relu -> f32 [16 x 256]. One wave per n-tile.
// ------------------------------------------------------------------
__global__ __launch_bounds__(32)
void fc2_wmma(const _Float16* __restrict__ h1, const float* __restrict__ fw,
              const float* __restrict__ bias, float* __restrict__ h2) {
    const int lane = threadIdx.x;
    const int nt = blockIdx.x;                          // 0..15
    const int g = lane >> 4, lm = lane & 15;
    const _Float16* ap = h1 + lm * 256;
    const float* bp = fw + (nt * 16 + lm) * 256;

    v8f acc = {};
    for (int c = 0; c < 8; ++c) {
        const int kb = c * 32;
        v8h alo = *reinterpret_cast<const v8h*>(ap + kb + 16 * g);
        v8h ahi = *reinterpret_cast<const v8h*>(ap + kb + 16 * g + 8);
        v16h a;
        for (int i = 0; i < 8; ++i) { a[i] = alo[i]; a[8 + i] = ahi[i]; }
        v16f bf = *reinterpret_cast<const v16f*>(bp + kb + 16 * g);
        v16h bb;
        for (int i = 0; i < 16; ++i) bb[i] = (_Float16)bf[i];
        acc = __builtin_amdgcn_wmma_f32_16x16x32_f16(
            false, a, false, bb, (short)0, acc, false, false);
    }
    for (int r = 0; r < 8; ++r) {
        const int m = g * 8 + r;
        h2[m * 256 + nt * 16 + lm] = fmaxf(acc[r] + bias[nt * 16 + lm], 0.f);
    }
}

__global__ __launch_bounds__(32)
void fc3_kernel(const float* __restrict__ h2, const float* __restrict__ fw3,
                const float* __restrict__ fb3, float* __restrict__ out) {
    const int m = threadIdx.x;
    if (m < 16) {
        float s = fb3[0];
        for (int k = 0; k < 256; ++k) s += h2[m * 256 + k] * fw3[k];
        out[m] = 1.f / (1.f + expf(-s));
    }
}

// ------------------------------------------------------------------
extern "C" void kernel_launch(void* const* d_in, const int* in_sizes, int n_in,
                              void* d_out, int out_size, void* d_ws, size_t ws_size,
                              hipStream_t stream) {
    (void)in_sizes; (void)n_in; (void)out_size; (void)ws_size;
    const float* x   = (const float*)d_in[0];
    const float* w1  = (const float*)d_in[1];  const float* b1 = (const float*)d_in[2];
    const float* w2  = (const float*)d_in[3];  const float* b2 = (const float*)d_in[4];
    const float* w3  = (const float*)d_in[5];  const float* b3 = (const float*)d_in[6];
    const float* w4  = (const float*)d_in[7];  const float* b4 = (const float*)d_in[8];
    const float* w5  = (const float*)d_in[9];  const float* b5 = (const float*)d_in[10];
    const float* w6  = (const float*)d_in[11]; const float* b6 = (const float*)d_in[12];
    const float* fw1 = (const float*)d_in[13]; const float* fb1 = (const float*)d_in[14];
    const float* fw2 = (const float*)d_in[15]; const float* fb2 = (const float*)d_in[16];
    const float* fw3 = (const float*)d_in[17]; const float* fb3 = (const float*)d_in[18];
    float* out = (float*)d_out;

    char* ws = (char*)d_ws;
    size_t off = 0;
    auto alloc = [&](size_t bytes) -> void* {
        void* p = (void*)(ws + off);
        off += (bytes + 255) & ~(size_t)255;
        return p;
    };
    _Float16* wp2 = (_Float16*)alloc((size_t)32 * 32 * 25 * 2);
    _Float16* wp3 = (_Float16*)alloc((size_t)64 * 32 * 9 * 2);
    _Float16* wp4 = (_Float16*)alloc((size_t)64 * 64 * 25 * 2);
    _Float16* wp5 = (_Float16*)alloc((size_t)128 * 64 * 9 * 2);
    _Float16* wp6 = (_Float16*)alloc((size_t)128 * 128 * 25 * 2);
    float*    fc1acc = (float*)alloc(16 * 256 * 4);
    _Float16* h1  = (_Float16*)alloc(16 * 256 * 2);
    float*    h2  = (float*)alloc(16 * 256 * 4);
    _Float16* act0 = (_Float16*)alloc((size_t)16 * 3 * 65536 * 2);      // co-occurrence
    _Float16* bufA = (_Float16*)alloc((size_t)16 * 32 * 65536 * 2);     // 67MB
    _Float16* bufB = (_Float16*)alloc((size_t)16 * 32 * 65536 * 2);     // 67MB

    // Weight packing (tiny, once per launch)
    pack_weights<<<(32 * 32 * 25 + 255) / 256, 256, 0, stream>>>(w2, wp2, 32, 32, 5, 32 * 32 * 25);
    pack_weights<<<(64 * 32 * 9 + 255) / 256, 256, 0, stream>>>(w3, wp3, 64, 32, 3, 64 * 32 * 9);
    pack_weights<<<(64 * 64 * 25 + 255) / 256, 256, 0, stream>>>(w4, wp4, 64, 64, 5, 64 * 64 * 25);
    pack_weights<<<(128 * 64 * 9 + 255) / 256, 256, 0, stream>>>(w5, wp5, 128, 64, 3, 128 * 64 * 9);
    pack_weights<<<(128 * 128 * 25 + 255) / 256, 256, 0, stream>>>(w6, wp6, 128, 128, 5, 128 * 128 * 25);

    // Stage 0: co-occurrence (48 planes, 256KB LDS histogram each)
    cooc_kernel<<<48, 1024, 256 * 1024, stream>>>(x, act0);

    // conv1 3->32 3x3 relu (direct, planar -> blocked)
    conv1_kernel<<<dim3(32, 256, 16), 256, 0, stream>>>(act0, w1, b1, bufA);

    // conv2 32->32 5x5 (WMMA) ; pool -> [16][1][128][128][32]
    conv_wmma<32, 32, 5, 2, false><<<dim3(16 * 2, 256, 16), 32, 0, stream>>>(bufA, wp2, b2, bufB, 256, 256);
    maxpool2_blk<false><<<(16 * 128 * 128 * 32 + 255) / 256, 256, 0, stream>>>(bufB, bufA, 1, 256, 256, 16 * 128 * 128 * 32);

    // conv3 32->64 3x3 relu
    conv_wmma<32, 64, 3, 1, true><<<dim3(8 * 4, 128, 16), 32, 0, stream>>>(bufA, wp3, b3, bufB, 128, 128);
    // conv4 64->64 5x5 ; pool -> [16][2][64][64][32]
    conv_wmma<64, 64, 5, 2, false><<<dim3(8 * 4, 128, 16), 32, 0, stream>>>(bufB, wp4, b4, bufA, 128, 128);
    maxpool2_blk<false><<<(16 * 2 * 64 * 64 * 32 + 255) / 256, 256, 0, stream>>>(bufA, bufB, 2, 128, 128, 16 * 2 * 64 * 64 * 32);

    // conv5 64->128 3x3 relu
    conv_wmma<64, 128, 3, 1, true><<<dim3(4 * 8, 64, 16), 32, 0, stream>>>(bufB, wp5, b5, bufA, 64, 64);
    // conv6 128->128 5x5 ; pool -> planar [16,128,32,32] for FC flatten
    conv_wmma<128, 128, 5, 2, false><<<dim3(4 * 8, 64, 16), 32, 0, stream>>>(bufA, wp6, b6, bufB, 64, 64);
    maxpool2_blk<true><<<(16 * 4 * 32 * 32 * 32 + 255) / 256, 256, 0, stream>>>(bufB, bufA, 4, 64, 64, 16 * 4 * 32 * 32 * 32);

    // FC head
    zero_f32<<<(16 * 256 + 255) / 256, 256, 0, stream>>>(fc1acc, 16 * 256);
    fc1_wmma<<<dim3(16, KSPLIT), 32, 0, stream>>>(bufA, fw1, fc1acc);
    fc1_post<<<(16 * 256 + 255) / 256, 256, 0, stream>>>(fc1acc, fb1, h1);
    fc2_wmma<<<16, 32, 0, stream>>>(h1, fw2, fb2, h2);
    fc3_kernel<<<1, 32, 0, stream>>>(h2, fw3, fb3, out);
}
